// WindowAttention_3066606649867
// MI455X (gfx1250) — matrix operand order repaired and checked by reading, not compiled, not run on hardware
//
#include <hip/hip_runtime.h>
#include <hip/hip_bf16.h>

// ---------------------------------------------------------------------------
// Swin window attention forward, MI455X (gfx1250), f16 WMMA pipeline.
// One 256-thread block (8 wave32) per window (49 tokens, C=128, 4 heads).
// All intermediates in LDS (~104KB -> 3 blocks/WGP); 4 GEMMs via
// v_wmma_f32_16x16x32_f16. Memory-bound: ~210MB HBM @ 23.3TB/s ~= 9us.
// ---------------------------------------------------------------------------

typedef _Float16 f16;
typedef __attribute__((ext_vector_type(16))) _Float16 v16h;
typedef __attribute__((ext_vector_type(8)))  _Float16 v8h;
typedef __attribute__((ext_vector_type(8)))  float    v8f;

#define N_TOK   49
#define DIMC    128
#define NHEAD   4
#define HDIM    32
#define SCALE_F 0.17677669529663687f   // 32^-0.5

// LDS row strides (halves). Multiples of 8 keep 16B alignment for ds_load_b128;
// +8 padding breaks the 64-bank power-of-two stride.
#define XS  136   // x / attn-out staging, 64 x 128 f16
#define QKS 264   // q|k staging, 64 x 256 f16
#define VTS 72    // v transposed, (4*32) x 64 f16
#define PS  72    // scores/probs per head, 64 x 64 f16

// --- WMMA fragment loaders (ISA 7.12.2 layouts) -----------------------------
// A (16x32 f16): lane m=lane&15, half h=lane>>4:
//   elems 0..7 = K[h*8 .. h*8+7], elems 8..15 = K[16+h*8 .. 16+h*8+7]
__device__ __forceinline__ v16h load_a(const f16* base, int stride, int row0, int kbase) {
    const int lane = threadIdx.x & 31;
    const f16* p = base + (size_t)(row0 + (lane & 15)) * stride + kbase + (lane >> 4) * 8;
    v8h lo = *(const v8h*)(p);
    v8h hi = *(const v8h*)(p + 16);
    return __builtin_shufflevector(lo, hi, 0,1,2,3,4,5,6,7,8,9,10,11,12,13,14,15);
}

// B (32x16 f16), stored K-major per column (B^T row-major):
// lane n=lane&15, half h=lane>>4: elems 0..7 = K[h*8 .. h*8+7], elems 8..15 = K[16+h*8 .. 16+h*8+7]
__device__ __forceinline__ v16h load_b(const f16* base, int stride, int col0, int kbase) {
    const int lane = threadIdx.x & 31;
    const f16* p = base + (size_t)(col0 + (lane & 15)) * stride + kbase + (lane >> 4) * 8;
    v8h lo = *(const v8h*)(p);
    v8h hi = *(const v8h*)(p + 16);
    return __builtin_shufflevector(lo, hi, 0,1,2,3,4,5,6,7,8,9,10,11,12,13,14,15);
}

// ---------------------------------------------------------------------------
// Prep kernels: f32->f16 weight transpose; padded bias gather; padded mask.
// ---------------------------------------------------------------------------
__global__ void prep_weights(const float* __restrict__ qkv_w,
                             const float* __restrict__ proj_w,
                             f16* __restrict__ wqkv_t, f16* __restrict__ wproj_t) {
    int i = blockIdx.x * 256 + threadIdx.x;
    if (i < 384 * 128) {                       // wqkv_t[c][k] = qkv_w[k][c]
        int c = i >> 7, k = i & 127;
        wqkv_t[c * 128 + k] = (f16)qkv_w[k * 384 + c];
    } else {
        i -= 384 * 128;
        if (i < 128 * 128) {                   // wproj_t[c][k] = proj_w[k][c]
            int c = i >> 7, k = i & 127;
            wproj_t[c * 128 + k] = (f16)proj_w[k * 128 + c];
        }
    }
}

__global__ void prep_bias(const float* __restrict__ table,
                          const int* __restrict__ ridx,
                          float* __restrict__ bias_pad) {   // [4][64][64]
    int i = blockIdx.x * 256 + threadIdx.x;
    if (i >= NHEAD * 64 * 64) return;
    int h = i >> 12, m = (i >> 6) & 63, c = i & 63;
    float v;
    if (c >= N_TOK)      v = -1e30f;   // padded key cols -> -inf before softmax
    else if (m >= N_TOK) v = 0.f;      // padded query rows: finite, discarded later
    else                 v = table[ridx[m * N_TOK + c] * NHEAD + h];
    bias_pad[i] = v;
}

__global__ void prep_mask(const float* __restrict__ mask,
                          float* __restrict__ mask_pad) {   // [64][64][64]
    int i = blockIdx.x * 256 + threadIdx.x;
    if (i >= 64 * 64 * 64) return;
    int w = i >> 12, m = (i >> 6) & 63, c = i & 63;
    mask_pad[i] = (m < N_TOK && c < N_TOK) ? mask[(w * N_TOK + m) * N_TOK + c] : 0.f;
}

// ---------------------------------------------------------------------------
// Main kernel: one block per window.
// ---------------------------------------------------------------------------
__global__ __launch_bounds__(256)
void swin_window_attn(const float* __restrict__ x,
                      const float* __restrict__ qkv_b,
                      const float* __restrict__ proj_b,
                      const f16*   __restrict__ wqkv_t,
                      const f16*   __restrict__ wproj_t,
                      const float* __restrict__ bias_pad,
                      const float* __restrict__ mask_pad,
                      float* __restrict__ out) {
    __shared__ f16 x_lds[64 * XS];              // reused as o_lds after GEMM1
    __shared__ f16 qk_lds[64 * QKS];            // q: cols 0..127, k: cols 128..255
    __shared__ f16 vt_lds[NHEAD * HDIM * VTS];  // v transposed: [h*32+d][m]
    __shared__ f16 p_lds[NHEAD * 64 * PS];      // scores -> probs, per head
    f16* o_lds = x_lds;                         // x dead after GEMM1; same size

    const int b    = blockIdx.x;
    const int w    = b & 63;                    // mask window index = b % 64
    const int tid  = threadIdx.x;
    const int wave = tid >> 5;
    const int lane = tid & 31;
    const int hs   = lane >> 4;
    const int ln   = lane & 15;

    // ---- stage x -> LDS f16 (rows 49..63 zero) ----
    const float* xb = x + (size_t)b * N_TOK * DIMC;
    for (int i = tid; i < 64 * DIMC; i += 256) {
        int n = i >> 7, c = i & 127;
        float v = (n < N_TOK) ? xb[n * DIMC + c] : 0.f;
        x_lds[n * XS + c] = (f16)v;
    }
    __syncthreads();

    // ---- GEMM1a: q|k tiles (nt 0..15): 64 tiles / 8 waves, row-major store ----
    for (int t = wave; t < 64; t += 8) {
        int mt = t & 3, nt = t >> 2;
        v8f acc = {};
        #pragma unroll
        for (int kk = 0; kk < 4; ++kk) {
            v16h a  = load_a(x_lds, XS, mt * 16, kk * 32);
            v16h bm = load_b(wqkv_t, DIMC, nt * 16, kk * 32);
            acc = __builtin_amdgcn_wmma_f32_16x16x32_f16(false, a, false, bm,
                                                         (short)0, acc, false, false);
        }
        int col = nt * 16 + ln;
        float bc = qkv_b[col];
        int r0 = mt * 16 + 8 * hs;
        #pragma unroll
        for (int r = 0; r < 8; ++r)
            qk_lds[(r0 + r) * QKS + col] = (f16)(acc[r] + bc);
    }

    // ---- GEMM1b: v tiles (nt 16..23): 32 tiles / 8 waves, transposed store ----
    for (int t = 64 + wave; t < 96; t += 8) {
        int mt = t & 3, nt = t >> 2;
        v8f acc = {};
        #pragma unroll
        for (int kk = 0; kk < 4; ++kk) {
            v16h a  = load_a(x_lds, XS, mt * 16, kk * 32);
            v16h bm = load_b(wqkv_t, DIMC, nt * 16, kk * 32);
            acc = __builtin_amdgcn_wmma_f32_16x16x32_f16(false, a, false, bm,
                                                         (short)0, acc, false, false);
        }
        int col = nt * 16 + ln;
        float bc = qkv_b[col];
        int r0 = mt * 16 + 8 * hs;
        int hd = col - 256;                     // = h*32 + d
        #pragma unroll
        for (int r = 0; r < 8; ++r)             // 8 contiguous halves -> b128 store
            vt_lds[hd * VTS + r0 + r] = (f16)(acc[r] + bc);
    }
    __syncthreads();

    // ---- QK^T, all heads: S = q_h(64x32) @ k_h^T(32x64); 64 tiles / 8 waves ----
    // Scores stored scaled+biased+masked as f16 directly into p_lds.
    const float* maskw = mask_pad + (size_t)w * 64 * 64;
    for (int t = wave; t < 64; t += 8) {
        int h = t >> 4, mt = (t >> 2) & 3, nt = t & 3;
        v16h a  = load_a(qk_lds, QKS, mt * 16, h * HDIM);
        v16h bm = load_b(qk_lds, QKS, nt * 16, 128 + h * HDIM);
        v8f acc = {};
        acc = __builtin_amdgcn_wmma_f32_16x16x32_f16(false, a, false, bm,
                                                     (short)0, acc, false, false);
        const float* bp = bias_pad + h * 64 * 64;
        int cc = nt * 16 + ln;
        int r0 = mt * 16 + 8 * hs;
        #pragma unroll
        for (int r = 0; r < 8; ++r) {
            int m = r0 + r;
            float v = acc[r] * SCALE_F + bp[m * 64 + cc] + maskw[m * 64 + cc];
            p_lds[(h * 64 + m) * PS + cc] = (f16)v;
        }
    }
    __syncthreads();

    // ---- softmax: 256 rows (4 heads x 64), one thread per row, in-place ----
    {
        f16* row = p_lds + (size_t)tid * PS;    // tid = h*64 + m
        float mx = -3.4e38f;
        for (int c = 0; c < 64; ++c) mx = fmaxf(mx, (float)row[c]);
        float sum = 0.f;
        for (int c = 0; c < 64; ++c) {
            float e = __expf((float)row[c] - mx);
            row[c] = (f16)e;
            sum += e;
        }
        f16 inv = (f16)(1.f / sum);
        for (int c = 0; c < 64; ++c) row[c] = (f16)(row[c] * inv);
    }
    __syncthreads();

    // ---- PV, all heads: O_h = P(64x64) @ v_h(64x32); 32 tiles / 8 waves ----
    for (int t = wave; t < 32; t += 8) {
        int h = t >> 3, mt = (t >> 1) & 3, nt = t & 1;
        v8f acc = {};
        #pragma unroll
        for (int kk = 0; kk < 2; ++kk) {
            v16h a  = load_a(p_lds + h * 64 * PS, PS, mt * 16, kk * 32);
            v16h bm = load_b(vt_lds + h * HDIM * VTS, VTS, nt * 16, kk * 32);
            acc = __builtin_amdgcn_wmma_f32_16x16x32_f16(false, a, false, bm,
                                                         (short)0, acc, false, false);
        }
        int col = h * HDIM + nt * 16 + ln;
        int r0 = mt * 16 + 8 * hs;
        #pragma unroll
        for (int r = 0; r < 8; ++r)
            o_lds[(r0 + r) * XS + col] = (f16)acc[r];
    }
    __syncthreads();

    // ---- GEMM3: out = O(64x128) @ proj(128x128) + proj_b ----
    float* ob = out + (size_t)b * N_TOK * DIMC;
    for (int t = wave; t < 32; t += 8) {
        int mt = t & 3, nt = t >> 2;
        v8f acc = {};
        #pragma unroll
        for (int kk = 0; kk < 4; ++kk) {
            v16h a  = load_a(o_lds, XS, mt * 16, kk * 32);
            v16h bm = load_b(wproj_t, DIMC, nt * 16, kk * 32);
            acc = __builtin_amdgcn_wmma_f32_16x16x32_f16(false, a, false, bm,
                                                         (short)0, acc, false, false);
        }
        int col = nt * 16 + ln;
        float pb = proj_b[col];
        int r0 = mt * 16 + 8 * hs;
        #pragma unroll
        for (int r = 0; r < 8; ++r) {
            int m = r0 + r;
            if (m < N_TOK) ob[m * DIMC + col] = acc[r] + pb;
        }
    }
}

// ---------------------------------------------------------------------------
// Host launcher. Inputs: x, mask, qkv_w, qkv_b, proj_w, proj_b,
//                        rel_bias_table, rel_index
// ---------------------------------------------------------------------------
extern "C" void kernel_launch(void* const* d_in, const int* in_sizes, int n_in,
                              void* d_out, int out_size, void* d_ws, size_t ws_size,
                              hipStream_t stream) {
    const float* x      = (const float*)d_in[0];
    const float* mask   = (const float*)d_in[1];
    const float* qkv_w  = (const float*)d_in[2];
    const float* qkv_b  = (const float*)d_in[3];
    const float* proj_w = (const float*)d_in[4];
    const float* proj_b = (const float*)d_in[5];
    const float* rtab   = (const float*)d_in[6];
    const int*   ridx   = (const int*)d_in[7];
    float* out = (float*)d_out;

    // workspace carve-up (all 16B aligned)
    char* ws = (char*)d_ws;
    f16*   wqkv_t   = (f16*)ws;                               // 384*128*2  =  98304 B
    f16*   wproj_t  = (f16*)(ws + 98304);                     // 128*128*2  =  32768 B
    float* bias_pad = (float*)(ws + 98304 + 32768);           // 4*64*64*4  =  65536 B
    float* mask_pad = (float*)(ws + 98304 + 32768 + 65536);   // 64*64*64*4 = 1048576 B

    prep_weights<<<(384 * 128 + 128 * 128 + 255) / 256, 256, 0, stream>>>(
        qkv_w, proj_w, wqkv_t, wproj_t);
    prep_bias<<<(NHEAD * 64 * 64 + 255) / 256, 256, 0, stream>>>(rtab, ridx, bias_pad);
    prep_mask<<<(64 * 64 * 64 + 255) / 256, 256, 0, stream>>>(mask, mask_pad);

    const int num_windows = in_sizes[0] / (N_TOK * DIMC);     // 4096
    swin_window_attn<<<num_windows, 256, 0, stream>>>(
        x, qkv_b, proj_b, wqkv_t, wproj_t, bias_pad, mask_pad, out);
}
